// Decoder_85091892068453
// MI455X (gfx1250) — matrix operand order repaired and checked by reading, not compiled, not run on hardware
//
#include <hip/hip_runtime.h>
#include <hip/hip_bf16.h>
#include <cstdint>

// ---------------------------------------------------------------------------
// Problem constants (match reference)
// ---------------------------------------------------------------------------
#define HD   1024            // hidden size
#define DD   80              // target/MFCC dim
#define DP   96              // D padded to multiple of 32 for WMMA K-loop
#define BB   32              // batch
#define SS   512             // encoder seq len
#define TT   512             // decoder steps
#define G3   (3 * HD)        // 3 gates

typedef __bf16 bf16;
typedef bf16  v16bf __attribute__((ext_vector_type(16)));
typedef bf16  v8bf  __attribute__((ext_vector_type(8)));
typedef float v8f   __attribute__((ext_vector_type(8)));

union BFrag { v16bf v; v8bf h[2]; };

// A fragment (16x32 bf16, M rows x K): lane L (<16): row, K = k0+{0..7, 16..23}
//                                      lane L (>=16): row, K = k0+{8..15, 24..31}
__device__ __forceinline__ v16bf load_afrag(const bf16* row, int k0, int kA) {
  BFrag f;
  f.h[0] = *(const v8bf*)(row + k0 + kA);
  f.h[1] = *(const v8bf*)(row + k0 + 16 + kA);
  return f.v;
}
// B fragment (32x16 bf16, K x N): lane L holds column n = L&15 with
// K = k0+{0..7, 16..23} (L<16) or k0+{8..15, 24..31} (L>=16), matching A.
// Weights are stored row-major per output neuron.
__device__ __forceinline__ v16bf load_bfrag(const bf16* wrow, int k0, int kB) {
  BFrag f;
  f.h[0] = *(const v8bf*)(wrow + k0 + kB / 2);
  f.h[1] = *(const v8bf*)(wrow + k0 + kB / 2 + 16);
  return f.v;
}

__device__ __forceinline__ v8f wmma_bf16(v16bf a, v16bf b, v8f c) {
  return __builtin_amdgcn_wmma_f32_16x16x32_bf16(
      /*neg_a=*/false, a, /*neg_b=*/false, b,
      /*c_mod=*/(short)0, c, /*reuse_a=*/false, /*reuse_b=*/false);
}

__device__ __forceinline__ float sigmoidf_(float x) {
  return 1.0f / (1.0f + __expf(-x));
}

// ---------------------------------------------------------------------------
// One-time prep: convert weights to bf16 (W_ih zero-padded D=80 -> 96)
// ---------------------------------------------------------------------------
__global__ void k_prep(const float* __restrict__ Whh, const float* __restrict__ Wih,
                       const float* __restrict__ Wc,
                       bf16* __restrict__ Whh_b, bf16* __restrict__ Wih_b,
                       bf16* __restrict__ Wc_b) {
  size_t i = (size_t)blockIdx.x * 256 + threadIdx.x;
  const size_t n1 = (size_t)G3 * HD;     // W_hh
  const size_t n2 = (size_t)G3 * DP;     // W_ih padded
  const size_t n3 = (size_t)HD * 2 * HD; // W_c
  if (i < n1) { Whh_b[i] = (bf16)Whh[i]; return; }
  i -= n1;
  if (i < n2) {
    size_t r = i / DP, d = i - r * DP;
    Wih_b[i] = (bf16)((d < DD) ? Wih[r * DD + d] : 0.0f);
    return;
  }
  i -= n2;
  if (i < n3) Wc_b[i] = (bf16)Wc[i];
}

__global__ void k_init(const float* __restrict__ h0, float* __restrict__ hf,
                       bf16* __restrict__ hb) {
  int i = blockIdx.x * 256 + threadIdx.x;   // B*H = 32768 threads
  float v = h0[i];
  hf[i] = v;
  hb[i] = (bf16)v;
}

// ---------------------------------------------------------------------------
// Fused GRU cell: gx + gh via WMMA (bf16, f32 accum) + elementwise update.
// Grid: 16 blocks (64 h-columns each) x 256 threads (8 waves = 2 Mtiles x 4 Ntiles)
// ---------------------------------------------------------------------------
__global__ __launch_bounds__(256) void k_gru(
    const float* __restrict__ tgt,                      // [B,T,D]
    const float* __restrict__ bih, const float* __restrict__ bhh,
    const bf16* __restrict__ Whh,                       // [3H, H] bf16
    const bf16* __restrict__ Wih,                       // [3H, DP] bf16 (padded)
    const float* __restrict__ hpf, const bf16* __restrict__ hpb,  // h_{t-1}
    float* __restrict__ hnf, bf16* __restrict__ hnb,    // h_t
    int t) {
  __shared__ __align__(16) bf16 xs[BB * DP];            // teacher-forced input, padded
  int tid = threadIdx.x;
  for (int i = tid; i < BB * DP; i += 256) {
    int bb = i / DP, d = i - bb * DP;
    float v = 0.0f;
    if (t > 0 && d < DD) v = tgt[((size_t)bb * TT + (t - 1)) * DD + d];
    xs[i] = (bf16)v;
  }
  __syncthreads();

  int wave = tid >> 5, lane = tid & 31, lm = lane & 15;
  int I  = (wave >> 2) * 16;                  // M tile base (0 or 16)
  int Jb = blockIdx.x * 64 + (wave & 3) * 16; // N tile base within H
  int kA = (lane < 16) ? 0 : 8;
  int kB = (lane < 16) ? 0 : 16;
  int mrow = I + lm;
  int ncol = Jb + lm;

  v8f ar = {}, az = {}, anh = {}, anx = {};

  // ---- gh contributions: K = H over previous hidden state ----
  const bf16* hrow = hpb + (size_t)mrow * HD;
  const bf16* wr = Whh + (size_t)(0 * HD + ncol) * HD;
  const bf16* wz = Whh + (size_t)(1 * HD + ncol) * HD;
  const bf16* wn = Whh + (size_t)(2 * HD + ncol) * HD;
  for (int k0 = 0; k0 < HD; k0 += 32) {
    v16bf a = load_afrag(hrow, k0, kA);
    ar  = wmma_bf16(a, load_bfrag(wr, k0, kB), ar);
    az  = wmma_bf16(a, load_bfrag(wz, k0, kB), az);
    anh = wmma_bf16(a, load_bfrag(wn, k0, kB), anh);
  }
  // ---- gx contributions: K = DP over teacher-forced input (LDS A operand) ----
  const bf16* xrow = xs + mrow * DP;
  const bf16* vr = Wih + (size_t)(0 * HD + ncol) * DP;
  const bf16* vz = Wih + (size_t)(1 * HD + ncol) * DP;
  const bf16* vn = Wih + (size_t)(2 * HD + ncol) * DP;
  for (int k0 = 0; k0 < DP; k0 += 32) {
    v16bf a = load_afrag(xrow, k0, kA);
    ar  = wmma_bf16(a, load_bfrag(vr, k0, kB), ar);
    az  = wmma_bf16(a, load_bfrag(vz, k0, kB), az);
    anx = wmma_bf16(a, load_bfrag(vn, k0, kB), anx);
  }

  // ---- GRU elementwise update; lane holds col ncol, rows rbase..rbase+7 ----
  float br_ = bih[ncol] + bhh[ncol];
  float bz_ = bih[HD + ncol] + bhh[HD + ncol];
  float bxn = bih[2 * HD + ncol];
  float bhn = bhh[2 * HD + ncol];
  int rbase = I + ((lane < 16) ? 0 : 8);
#pragma unroll
  for (int i = 0; i < 8; i++) {
    int row = rbase + i;
    float r = sigmoidf_(ar[i] + br_);
    float z = sigmoidf_(az[i] + bz_);
    float n = tanhf(anx[i] + bxn + r * (anh[i] + bhn));
    float hp = hpf[row * HD + ncol];
    float hv = (1.0f - z) * n + z * hp;
    hnf[row * HD + ncol] = hv;
    hnb[row * HD + ncol] = (bf16)hv;
  }
}

// ---------------------------------------------------------------------------
// Attention scores: scores[b,s] = enc[b,s,:] . h[b,:]   (one wave per (b,s))
// ---------------------------------------------------------------------------
__global__ __launch_bounds__(256) void k_scores(const float* __restrict__ enc,
                                                const float* __restrict__ h,
                                                float* __restrict__ scores) {
  int gw = (blockIdx.x * 256 + threadIdx.x) >> 5;  // global wave id = b*S+s
  int lane = threadIdx.x & 31;
  int b = gw >> 9, s = gw & (SS - 1);
  const float4* e  = (const float4*)(enc + ((size_t)b * SS + s) * HD);
  const float4* hv = (const float4*)(h + (size_t)b * HD);
  float acc = 0.0f;
  for (int i = lane; i < HD / 4; i += 32) {
    float4 ev = e[i], hh = hv[i];
    acc += ev.x * hh.x + ev.y * hh.y + ev.z * hh.z + ev.w * hh.w;
  }
  for (int off = 16; off; off >>= 1) acc += __shfl_xor(acc, off, 32);
  if (lane == 0) scores[b * SS + s] = acc;
}

// ---------------------------------------------------------------------------
// Softmax over S per batch row (one 512-thread block per b)
// ---------------------------------------------------------------------------
__global__ __launch_bounds__(512) void k_softmax(const float* __restrict__ scores,
                                                 float* __restrict__ attn) {
  __shared__ float red[16];
  __shared__ float smax, ssum;
  int b = blockIdx.x, tid = threadIdx.x;
  float v = scores[b * SS + tid];
  float m = v;
  for (int off = 16; off; off >>= 1) m = fmaxf(m, __shfl_xor(m, off, 32));
  if ((tid & 31) == 0) red[tid >> 5] = m;
  __syncthreads();
  if (tid == 0) {
    float mm = red[0];
    for (int i = 1; i < 16; i++) mm = fmaxf(mm, red[i]);
    smax = mm;
  }
  __syncthreads();
  float e = __expf(v - smax);
  float s = e;
  for (int off = 16; off; off >>= 1) s += __shfl_xor(s, off, 32);
  if ((tid & 31) == 0) red[tid >> 5] = s;
  __syncthreads();
  if (tid == 0) {
    float tsum = 0.0f;
    for (int i = 0; i < 16; i++) tsum += red[i];
    ssum = tsum;
  }
  __syncthreads();
  attn[b * SS + tid] = e / ssum;
}

// ---------------------------------------------------------------------------
// Context: ctx[b,h] = sum_s attn[b,s] * enc[b,s,h]. Grid (B, H/256) x 256.
// ---------------------------------------------------------------------------
__global__ __launch_bounds__(256) void k_ctx(const float* __restrict__ enc,
                                             const float* __restrict__ attn,
                                             bf16* __restrict__ ctxb) {
  __shared__ float a[SS];
  int b = blockIdx.x;
  int h = blockIdx.y * 256 + threadIdx.x;
  for (int i = threadIdx.x; i < SS; i += 256) a[i] = attn[b * SS + i];
  __syncthreads();
  const float* e = enc + (size_t)b * SS * HD + h;
  float acc = 0.0f;
#pragma unroll 4
  for (int s = 0; s < SS; s++) acc = fmaf(a[s], e[(size_t)s * HD], acc);
  ctxb[b * HD + h] = (bf16)acc;
}

// ---------------------------------------------------------------------------
// Concat linear: u = tanh([h | ctx] @ W_c^T + b_c) via WMMA, K = 2H.
// Grid: 16 blocks x 256 threads (same tiling as k_gru).
// ---------------------------------------------------------------------------
__global__ __launch_bounds__(256) void k_concat(
    const bf16* __restrict__ hnb, const bf16* __restrict__ ctxb,
    const bf16* __restrict__ Wc, const float* __restrict__ bc,
    float* __restrict__ u) {
  int tid = threadIdx.x;
  int wave = tid >> 5, lane = tid & 31, lm = lane & 15;
  int I  = (wave >> 2) * 16;
  int Jb = blockIdx.x * 64 + (wave & 3) * 16;
  int kA = (lane < 16) ? 0 : 8;
  int kB = (lane < 16) ? 0 : 16;
  int mrow = I + lm, ncol = Jb + lm;

  v8f acc = {};
  const bf16* ah = hnb + (size_t)mrow * HD;
  const bf16* ac = ctxb + (size_t)mrow * HD;
  const bf16* wrow = Wc + (size_t)ncol * (2 * HD);
  for (int k0 = 0; k0 < HD; k0 += 32)
    acc = wmma_bf16(load_afrag(ah, k0, kA), load_bfrag(wrow, k0, kB), acc);
  for (int k0 = 0; k0 < HD; k0 += 32)
    acc = wmma_bf16(load_afrag(ac, k0, kA), load_bfrag(wrow + HD, k0, kB), acc);

  float b0 = bc[ncol];
  int rbase = I + ((lane < 16) ? 0 : 8);
#pragma unroll
  for (int i = 0; i < 8; i++)
    u[(rbase + i) * HD + ncol] = tanhf(acc[i] + b0);
}

// ---------------------------------------------------------------------------
// Output linear + length mask: out[b,t,d] = (t < len[b]) ? u[b,:].W_o[d,:]+b_o : 0
// One wave per (b,d): 2560 waves = 320 blocks x 8 waves.
// ---------------------------------------------------------------------------
__global__ __launch_bounds__(256) void k_out(const float* __restrict__ u,
                                             const float* __restrict__ Wo,
                                             const float* __restrict__ bo,
                                             const int* __restrict__ tlen,
                                             float* __restrict__ out, int t) {
  int gw = (blockIdx.x * 256 + threadIdx.x) >> 5;
  int lane = threadIdx.x & 31;
  int b = gw / DD, d = gw - b * DD;
  const float4* uv = (const float4*)(u + (size_t)b * HD);
  const float4* wv = (const float4*)(Wo + (size_t)d * HD);
  float acc = 0.0f;
  for (int i = lane; i < HD / 4; i += 32) {
    float4 a = uv[i], w = wv[i];
    acc += a.x * w.x + a.y * w.y + a.z * w.z + a.w * w.w;
  }
  for (int off = 16; off; off >>= 1) acc += __shfl_xor(acc, off, 32);
  if (lane == 0) {
    float val = acc + bo[d];
    out[((size_t)b * TT + t) * DD + d] = (t < tlen[b]) ? val : 0.0f;
  }
}

// ---------------------------------------------------------------------------
// Host-side orchestration
// ---------------------------------------------------------------------------
extern "C" void kernel_launch(void* const* d_in, const int* in_sizes, int n_in,
                              void* d_out, int out_size, void* d_ws, size_t ws_size,
                              hipStream_t stream) {
  const float* enc  = (const float*)d_in[0];   // [B,S,H]
  const float* h0   = (const float*)d_in[1];   // [1,B,H]
  const float* tgt  = (const float*)d_in[2];   // [B,T,D]
  const int*   tlen = (const int*)  d_in[3];   // [B]
  const float* Wih  = (const float*)d_in[4];   // [3H,D]
  const float* Whh  = (const float*)d_in[5];   // [3H,H]
  const float* bih  = (const float*)d_in[6];
  const float* bhh  = (const float*)d_in[7];
  const float* Wc   = (const float*)d_in[8];   // [H,2H]
  const float* bc   = (const float*)d_in[9];
  const float* Wo   = (const float*)d_in[10];  // [D,H]
  const float* bo   = (const float*)d_in[11];
  float* out = (float*)d_out;                  // [B,T,D]

  // workspace carving (~11.7 MB)
  char* w = (char*)d_ws;
  size_t o = 0;
  float* h_f[2]; bf16* h_b[2];
  h_f[0] = (float*)(w + o); o += (size_t)BB * HD * 4;
  h_f[1] = (float*)(w + o); o += (size_t)BB * HD * 4;
  h_b[0] = (bf16*)(w + o);  o += (size_t)BB * HD * 2;
  h_b[1] = (bf16*)(w + o);  o += (size_t)BB * HD * 2;
  float* scores = (float*)(w + o); o += (size_t)BB * SS * 4;
  float* attn   = (float*)(w + o); o += (size_t)BB * SS * 4;
  bf16*  ctx_b  = (bf16*)(w + o);  o += (size_t)BB * HD * 2;
  float* u      = (float*)(w + o); o += (size_t)BB * HD * 4;
  bf16* Whh_b = (bf16*)(w + o); o += (size_t)G3 * HD * 2;
  bf16* Wih_b = (bf16*)(w + o); o += (size_t)G3 * DP * 2;
  bf16* Wc_b  = (bf16*)(w + o); o += (size_t)HD * 2 * HD * 2;
  (void)ws_size; (void)in_sizes; (void)n_in; (void)out_size;

  // one-time (per-call) weight conversion + state init
  {
    size_t total = (size_t)G3 * HD + (size_t)G3 * DP + (size_t)HD * 2 * HD;
    int blocks = (int)((total + 255) / 256);
    k_prep<<<blocks, 256, 0, stream>>>(Whh, Wih, Wc, Whh_b, Wih_b, Wc_b);
    k_init<<<(BB * HD) / 256, 256, 0, stream>>>(h0, h_f[0], h_b[0]);
  }

  for (int t = 0; t < TT; t++) {
    int cur = t & 1;       // h_{t-1} buffer
    int nxt = cur ^ 1;     // h_t buffer
    k_gru<<<HD / 64, 256, 0, stream>>>(tgt, bih, bhh, Whh_b, Wih_b,
                                       h_f[cur], h_b[cur], h_f[nxt], h_b[nxt], t);
    k_scores<<<(BB * SS) / 8, 256, 0, stream>>>(enc, h_f[nxt], scores);
    k_softmax<<<BB, SS, 0, stream>>>(scores, attn);
    k_ctx<<<dim3(BB, HD / 256), 256, 0, stream>>>(enc, attn, ctx_b);
    k_concat<<<HD / 64, 256, 0, stream>>>(h_b[nxt], ctx_b, Wc_b, bc, u);
    k_out<<<(BB * DD) / 8, 256, 0, stream>>>(u, Wo, bo, tlen, out, t);
  }
}
